// DVBF_21818433863855
// MI455X (gfx1250) — matrix operand order repaired and checked by reading, not compiled, not run on hardware
//
#include <hip/hip_runtime.h>
#include <hip/hip_bf16.h>
#include <math.h>

#define DZC   3
#define DBC   3
#define HID   128
#define XDIM  4096
#define EH    512
#define BATCH 128
#define TSEQ  64
#define DT_STEP 0.05f

typedef __attribute__((ext_vector_type(16))) __bf16 v16bf;
typedef __attribute__((ext_vector_type(8)))  __bf16 v8bf;
typedef __attribute__((ext_vector_type(8)))  float  v8f;

__device__ __forceinline__ float sigmoidf_(float x){ return 1.0f / (1.0f + __expf(-x)); }

// Build the 16x32 bf16 A fragment from its two 8-element K-chunks.
// Contiguous layout: lanes 0-15 hold K in kk..kk+15, lanes 16-31 hold kk+16..kk+31.
__device__ __forceinline__ v16bf mk_a(v8bf c0, v8bf c1){
  v16bf a;
  #pragma unroll
  for (int i = 0; i < 8; i++){ a[i] = c0[i]; a[i + 8] = c1[i]; }
  return a;
}

// ---------------------------------------------------------------------------
// Weight prep: f32 (K x N) -> bf16 transposed (N x K), and plain f32->bf16.
// ---------------------------------------------------------------------------
__global__ void k_transpose_bf16(const float* __restrict__ W, __bf16* __restrict__ Wt, int K, int N){
  int idx = blockIdx.x * 256 + threadIdx.x;
  if (idx >= K * N) return;
  int k = idx / N, n = idx - k * N;
  Wt[(long)n * K + k] = (__bf16)W[idx];
}

__global__ void k_conv_bf16(const float* __restrict__ W, __bf16* __restrict__ O, int n){
  int idx = blockIdx.x * 256 + threadIdx.x;
  if (idx >= n) return;
  O[idx] = (__bf16)W[idx];
}

// ---------------------------------------------------------------------------
// Templated bf16 WMMA GEMM:  C[M,N] = ACT(A[M,K] @ Bt[N,K]^T + bias)
//   AF32 : A is f32 (converted to bf16 in-flight) vs bf16
//   AMODE/CMODE==1 : remap row m=(t*128+b) <-> memory row (b*64+t)
//   ACT  : 0 none, 1 relu, 2 sigmoid
// One wave computes a 32x64 tile: 2 A-frags x 4 B-frags -> 8 WMMAs / K-step,
// halving B traffic vs a 16-row tile.  B panel prefetched one K-step ahead.
// ---------------------------------------------------------------------------
template<int AF32, int AMODE, int CMODE, int ACT>
__global__ void __launch_bounds__(256) k_wmma_gemm(
    const float*  __restrict__ Af,
    const __bf16* __restrict__ Ab,
    long a_off, long lda,
    const __bf16* __restrict__ Bt,
    const float*  __restrict__ bias,
    float* __restrict__ C, long c_off, long ldc,
    int M, int N, int K)
{
  const int lane = threadIdx.x & 31;
  const int wave = threadIdx.x >> 5;
  const int ntg  = N >> 6;
  const int total = (M >> 5) * ntg;
  int wt = blockIdx.x * 8 + wave;
  if (wt >= total) return;
  const int mt = wt / ntg, ng = wt - mt * ntg;
  const int m0 = mt << 5, n0 = ng << 6;
  const int lrow  = lane & 15;
  const int koffA = (lane >> 4) << 4;   // 0 or 16 (A contiguous layout)
  const int koffB = (lane >> 4) << 4;   // 0 or 16 (B column layout)

  long abase0, abase1;
  {
    const int r0 = m0 + lrow, r1 = m0 + 16 + lrow;
    if (AMODE == 1){
      abase0 = ((long)((r0 & 127) * TSEQ + (r0 >> 7))) * lda + a_off;
      abase1 = ((long)((r1 & 127) * TSEQ + (r1 >> 7))) * lda + a_off;
    } else {
      abase0 = (long)r0 * lda + a_off;
      abase1 = (long)r1 * lda + a_off;
    }
  }
  const long bb = (long)(n0 + lrow) * K + koffB;

  v8f acc[8];
  #pragma unroll
  for (int j = 0; j < 8; j++)
    #pragma unroll
    for (int i = 0; i < 8; i++) acc[j][i] = 0.0f;

  for (int kk = 0; kk < K; kk += 32){
    v16bf a0, a1;
    if (AF32){
      const float* ap0 = Af + abase0 + kk + koffA;
      const float* ap1 = Af + abase1 + kk + koffA;
      v8f c00 = *(const v8f*)ap0;
      v8f c01 = *(const v8f*)(ap0 + 8);
      v8f c10 = *(const v8f*)ap1;
      v8f c11 = *(const v8f*)(ap1 + 8);
      #pragma unroll
      for (int i = 0; i < 8; i++){
        a0[i] = (__bf16)c00[i]; a0[i + 8] = (__bf16)c01[i];
        a1[i] = (__bf16)c10[i]; a1[i + 8] = (__bf16)c11[i];
      }
    } else {
      const __bf16* ap0 = Ab + abase0 + kk + koffA;
      const __bf16* ap1 = Ab + abase1 + kk + koffA;
      a0 = mk_a(*(const v8bf*)ap0, *(const v8bf*)(ap0 + 8));
      a1 = mk_a(*(const v8bf*)ap1, *(const v8bf*)(ap1 + 8));
    }
    const __bf16* bp = Bt + bb + kk;
    v16bf b0 = *(const v16bf*)(bp);
    v16bf b1 = *(const v16bf*)(bp + 16L * K);
    v16bf b2 = *(const v16bf*)(bp + 32L * K);
    v16bf b3 = *(const v16bf*)(bp + 48L * K);
    __builtin_prefetch((const void*)(bp + 32), 0, 1);  // next K-step B panel
    acc[0] = __builtin_amdgcn_wmma_f32_16x16x32_bf16(false, a0, false, b0, (short)0, acc[0], false, false);
    acc[1] = __builtin_amdgcn_wmma_f32_16x16x32_bf16(false, a0, false, b1, (short)0, acc[1], false, false);
    acc[2] = __builtin_amdgcn_wmma_f32_16x16x32_bf16(false, a0, false, b2, (short)0, acc[2], false, false);
    acc[3] = __builtin_amdgcn_wmma_f32_16x16x32_bf16(false, a0, false, b3, (short)0, acc[3], false, false);
    acc[4] = __builtin_amdgcn_wmma_f32_16x16x32_bf16(false, a1, false, b0, (short)0, acc[4], false, false);
    acc[5] = __builtin_amdgcn_wmma_f32_16x16x32_bf16(false, a1, false, b1, (short)0, acc[5], false, false);
    acc[6] = __builtin_amdgcn_wmma_f32_16x16x32_bf16(false, a1, false, b2, (short)0, acc[6], false, false);
    acc[7] = __builtin_amdgcn_wmma_f32_16x16x32_bf16(false, a1, false, b3, (short)0, acc[7], false, false);
  }

  #pragma unroll
  for (int half = 0; half < 2; half++){
    const int rb = m0 + half * 16 + ((lane >> 4) << 3);
    #pragma unroll
    for (int j = 0; j < 4; j++){
      int col = n0 + j * 16 + lrow;
      float bv = bias[col];
      #pragma unroll
      for (int v = 0; v < 8; v++){
        float val = acc[half * 4 + j][v] + bv;
        if      (ACT == 1) val = fmaxf(val, 0.0f);
        else if (ACT == 2) val = sigmoidf_(val);
        int r = rb + v;
        long crow = (CMODE == 1) ? ((long)(r & 127) * TSEQ + (r >> 7)) : (long)r;
        C[crow * ldc + c_off + col] = val;
      }
    }
  }
}

// ---------------------------------------------------------------------------
// z = p0 + exp(p1)*eps, p = h_enc @ enc_w2 + enc_b2 (N=6, VALU dot).
// ---------------------------------------------------------------------------
__global__ void k_encode_z(const float* __restrict__ h_enc,
                           const float* __restrict__ enc_w2, const float* __restrict__ enc_b2,
                           const float* __restrict__ eps_z, float* __restrict__ z,
                           int M, int t_base)
{
  int idx = blockIdx.x * 256 + threadIdx.x;
  if (idx >= M * DZC) return;
  int m = idx / DZC, d = idx - m * DZC;
  const float* hr = h_enc + (long)m * EH;
  float p0 = enc_b2[2 * d], p1 = enc_b2[2 * d + 1];
  for (int k = 0; k < EH; k++){
    float hv = hr[k];
    p0 += hv * enc_w2[k * 6 + 2 * d];
    p1 += hv * enc_w2[k * 6 + 2 * d + 1];
  }
  int b = m & 127, t = t_base + (m >> 7);
  z[idx] = p0 + __expf(p1) * eps_z[((long)b * TSEQ + t) * DZC + d];
}

// ---------------------------------------------------------------------------
// Sequential LSTM scan, steps 0..62.  One workgroup (8 waves) keeps
// h (bf16, 32KB) + c (f32, 64KB) resident in LDS; gates staged in LDS bf16
// (128KB).  Per step the 128x512x128 gate GEMM runs on WMMA.
// ---------------------------------------------------------------------------
__global__ void __launch_bounds__(256) k_seq(
    const float* __restrict__ z, const float* __restrict__ u,
    const float* __restrict__ eps_beta,
    const __bf16* __restrict__ whh_bf,
    const float* __restrict__ w_ih, const float* __restrict__ b_ih, const float* __restrict__ b_hh,
    const float* __restrict__ fc_w, const float* __restrict__ fc_b,
    const float* __restrict__ t1w, const float* __restrict__ t1b,
    const float* __restrict__ t2w, const float* __restrict__ t2b,
    const float* __restrict__ t3w, const float* __restrict__ t3b,
    const float* __restrict__ t4w, const float* __restrict__ t4b,
    float* __restrict__ z_next, float* __restrict__ h_out, float* __restrict__ c_out,
    int nsteps)
{
  __shared__ __align__(32) __bf16 h_bf[BATCH * HID];
  __shared__ __align__(32) __bf16 g_s[BATCH * 512];
  __shared__ float c_s[BATCH * HID];
  const int tid = threadIdx.x;
  const int lane = tid & 31, wave = tid >> 5;
  for (int e = tid; e < BATCH * HID; e += 256){ h_bf[e] = (__bf16)0.0f; c_s[e] = 0.0f; }
  __syncthreads();

  const int lrow  = lane & 15;
  const int koffA = (lane >> 4) << 4;
  const int koffB = (lane >> 4) << 4;
  const int m0 = wave << 4;

  for (int t = 0; t < nsteps; t++){
    // ---- Phase A: gates(partial) = h @ w_hh^T via WMMA, into LDS ----
    for (int nc = 0; nc < 8; nc++){
      const int n0 = nc << 6;
      v8f acc[4];
      #pragma unroll
      for (int j = 0; j < 4; j++)
        #pragma unroll
        for (int i = 0; i < 8; i++) acc[j][i] = 0.0f;
      #pragma unroll
      for (int kk = 0; kk < HID; kk += 32){
        const __bf16* ap = &h_bf[(m0 + lrow) * HID + kk + koffA];
        v16bf a = mk_a(*(const v8bf*)ap, *(const v8bf*)(ap + 8));
        const __bf16* bp = whh_bf + (long)(n0 + lrow) * HID + kk + koffB;
        v16bf b0 = *(const v16bf*)(bp);
        v16bf b1 = *(const v16bf*)(bp + 16 * HID);
        v16bf b2 = *(const v16bf*)(bp + 32 * HID);
        v16bf b3 = *(const v16bf*)(bp + 48 * HID);
        acc[0] = __builtin_amdgcn_wmma_f32_16x16x32_bf16(false, a, false, b0, (short)0, acc[0], false, false);
        acc[1] = __builtin_amdgcn_wmma_f32_16x16x32_bf16(false, a, false, b1, (short)0, acc[1], false, false);
        acc[2] = __builtin_amdgcn_wmma_f32_16x16x32_bf16(false, a, false, b2, (short)0, acc[2], false, false);
        acc[3] = __builtin_amdgcn_wmma_f32_16x16x32_bf16(false, a, false, b3, (short)0, acc[3], false, false);
      }
      const int rb = m0 + ((lane >> 4) << 3);
      #pragma unroll
      for (int j = 0; j < 4; j++){
        int col = n0 + j * 16 + lrow;
        #pragma unroll
        for (int v = 0; v < 8; v++) g_s[(rb + v) * 512 + col] = (__bf16)acc[j][v];
      }
    }
    __syncthreads();

    // ---- Phase B: LSTM cell elementwise ----
    const float* zt = z + (long)t * BATCH * DZC;
    for (int e = tid; e < BATCH * HID; e += 256){
      int b = e >> 7, k = e & 127;
      float z0 = zt[b * 3], z1 = zt[b * 3 + 1], z2 = zt[b * 3 + 2];
      float gi = (float)g_s[b * 512 + k]       + b_ih[k]       + b_hh[k]       + z0 * w_ih[(k)       * 3] + z1 * w_ih[(k)       * 3 + 1] + z2 * w_ih[(k)       * 3 + 2];
      float gf = (float)g_s[b * 512 + 128 + k] + b_ih[128 + k] + b_hh[128 + k] + z0 * w_ih[(128 + k) * 3] + z1 * w_ih[(128 + k) * 3 + 1] + z2 * w_ih[(128 + k) * 3 + 2];
      float gg = (float)g_s[b * 512 + 256 + k] + b_ih[256 + k] + b_hh[256 + k] + z0 * w_ih[(256 + k) * 3] + z1 * w_ih[(256 + k) * 3 + 1] + z2 * w_ih[(256 + k) * 3 + 2];
      float go = (float)g_s[b * 512 + 384 + k] + b_ih[384 + k] + b_hh[384 + k] + z0 * w_ih[(384 + k) * 3] + z1 * w_ih[(384 + k) * 3 + 1] + z2 * w_ih[(384 + k) * 3 + 2];
      float cn = sigmoidf_(gf) * c_s[e] + sigmoidf_(gi) * tanhf(gg);
      float hn = sigmoidf_(go) * tanhf(cn);
      c_s[e] = cn;
      h_bf[e] = (__bf16)hn;
    }
    __syncthreads();

    // ---- Phase C: beta, transition MLP, z_next (one thread per batch row) ----
    if (tid < BATCH){
      const int b = tid;
      float zb0 = zt[b * 3], zb1 = zt[b * 3 + 1], zb2 = zt[b * 3 + 2];
      float p[6];
      #pragma unroll
      for (int d = 0; d < 6; d++) p[d] = fc_b[d];
      for (int k = 0; k < HID; k++){
        float hv = (float)h_bf[b * HID + k];
        #pragma unroll
        for (int d = 0; d < 6; d++) p[d] += hv * fc_w[k * 6 + d];
      }
      float inp[7];
      inp[0] = zb0; inp[1] = zb1; inp[2] = zb2;
      inp[3] = u[(long)b * TSEQ + t];
      #pragma unroll
      for (int d = 0; d < 3; d++)
        inp[4 + d] = p[2 * d] + __expf(p[2 * d + 1]) * eps_beta[((long)b * TSEQ + t) * 3 + d];
      float h1[32];
      #pragma unroll
      for (int j = 0; j < 32; j++){
        float s = t1b[j];
        #pragma unroll
        for (int i = 0; i < 7; i++) s += inp[i] * t1w[i * 32 + j];
        h1[j] = fmaxf(s, 0.0f);
      }
      float h2[16];
      #pragma unroll
      for (int j = 0; j < 16; j++){
        float s = t2b[j];
        #pragma unroll
        for (int i = 0; i < 32; i++) s += h1[i] * t2w[i * 16 + j];
        h2[j] = fmaxf(s, 0.0f);
      }
      float h3[8];
      #pragma unroll
      for (int j = 0; j < 8; j++){
        float s = t3b[j];
        #pragma unroll
        for (int i = 0; i < 16; i++) s += h2[i] * t3w[i * 8 + j];
        h3[j] = fmaxf(s, 0.0f);
      }
      float zn[3] = {zb0, zb1, zb2};
      #pragma unroll
      for (int d = 0; d < 3; d++){
        float s = t4b[d];
        #pragma unroll
        for (int i = 0; i < 8; i++) s += h3[i] * t4w[i * 3 + d];
        zn[d] += s * DT_STEP;
      }
      float* zo = z_next + ((long)t * BATCH + b) * 3;
      zo[0] = zn[0]; zo[1] = zn[1]; zo[2] = zn[2];
    }
    __syncthreads();
  }
  for (int e = tid; e < BATCH * HID; e += 256){ h_out[e] = (float)h_bf[e]; c_out[e] = c_s[e]; }
}

// ---------------------------------------------------------------------------
// hdec = relu(z_next @ dec_w1 + dec_b1)  (K=3 -> VALU), stored bf16 for WMMA.
// ---------------------------------------------------------------------------
__global__ void k_hdec(const float* __restrict__ zn, const float* __restrict__ dec_w1,
                       const float* __restrict__ dec_b1, __bf16* __restrict__ hdec, int M)
{
  int idx = blockIdx.x * 256 + threadIdx.x;
  if (idx >= M * EH) return;
  int m = idx >> 9, j = idx & 511;
  float v = dec_b1[j] + zn[m * 3] * dec_w1[j] + zn[m * 3 + 1] * dec_w1[EH + j] + zn[m * 3 + 2] * dec_w1[2 * EH + j];
  hdec[idx] = (__bf16)fmaxf(v, 0.0f);
}

// ---------------------------------------------------------------------------
// Step-63 (mask==false branch): full gates via VALU dot (one-off, 8.4M MAC).
// ---------------------------------------------------------------------------
__global__ void k_step63_gates(const float* __restrict__ z63, const float* __restrict__ h,
                               const float* __restrict__ w_ih, const float* __restrict__ whh,
                               const float* __restrict__ b_ih, const float* __restrict__ b_hh,
                               float* __restrict__ gates)
{
  int idx = blockIdx.x * 256 + threadIdx.x;
  if (idx >= BATCH * 512) return;
  int b = idx >> 9, j = idx & 511;
  float g = b_ih[j] + b_hh[j]
          + z63[b * 3] * w_ih[j * 3] + z63[b * 3 + 1] * w_ih[j * 3 + 1] + z63[b * 3 + 2] * w_ih[j * 3 + 2];
  const float* hr = h + b * HID;
  const float* wr = whh + (long)j * HID;
  float s = 0.0f;
  for (int k = 0; k < HID; k++) s += hr[k] * wr[k];
  gates[idx] = g + s;
}

__global__ void __launch_bounds__(512) k_step63_finish(
    const float* __restrict__ gates, const float* __restrict__ c_in, const float* __restrict__ z63,
    const float* __restrict__ eps_beta,
    const float* __restrict__ fc_w, const float* __restrict__ fc_b,
    const float* __restrict__ t1w, const float* __restrict__ t1b,
    const float* __restrict__ t2w, const float* __restrict__ t2b,
    const float* __restrict__ t3w, const float* __restrict__ t3b,
    const float* __restrict__ t4w, const float* __restrict__ t4b,
    float* __restrict__ z_next63)
{
  __shared__ float h_s[BATCH * HID];
  int tid = threadIdx.x;
  for (int e = tid; e < BATCH * HID; e += 512){
    int b = e >> 7, k = e & 127;
    float gi = gates[b * 512 + k], gf = gates[b * 512 + 128 + k];
    float gg = gates[b * 512 + 256 + k], go = gates[b * 512 + 384 + k];
    float cn = sigmoidf_(gf) * c_in[e] + sigmoidf_(gi) * tanhf(gg);
    h_s[e] = sigmoidf_(go) * tanhf(cn);
  }
  __syncthreads();
  if (tid < BATCH){
    const int b = tid, t = 63;
    float zb0 = z63[b * 3], zb1 = z63[b * 3 + 1], zb2 = z63[b * 3 + 2];
    float p[6];
    #pragma unroll
    for (int d = 0; d < 6; d++) p[d] = fc_b[d];
    for (int k = 0; k < HID; k++){
      float hv = h_s[b * HID + k];
      #pragma unroll
      for (int d = 0; d < 6; d++) p[d] += hv * fc_w[k * 6 + d];
    }
    float inp[7];
    inp[0] = zb0; inp[1] = zb1; inp[2] = zb2;
    inp[3] = 0.0f;  // u masked off at step 63
    #pragma unroll
    for (int d = 0; d < 3; d++)
      inp[4 + d] = p[2 * d] + __expf(p[2 * d + 1]) * eps_beta[((long)b * TSEQ + t) * 3 + d];
    float h1[32];
    #pragma unroll
    for (int j = 0; j < 32; j++){
      float s = t1b[j];
      #pragma unroll
      for (int i = 0; i < 7; i++) s += inp[i] * t1w[i * 32 + j];
      h1[j] = fmaxf(s, 0.0f);
    }
    float h2[16];
    #pragma unroll
    for (int j = 0; j < 16; j++){
      float s = t2b[j];
      #pragma unroll
      for (int i = 0; i < 32; i++) s += h1[i] * t2w[i * 16 + j];
      h2[j] = fmaxf(s, 0.0f);
    }
    float h3[8];
    #pragma unroll
    for (int j = 0; j < 8; j++){
      float s = t3b[j];
      #pragma unroll
      for (int i = 0; i < 16; i++) s += h2[i] * t3w[i * 8 + j];
      h3[j] = fmaxf(s, 0.0f);
    }
    float zn[3] = {zb0, zb1, zb2};
    #pragma unroll
    for (int d = 0; d < 3; d++){
      float s = t4b[d];
      #pragma unroll
      for (int i = 0; i < 8; i++) s += h3[i] * t4w[i * 3 + d];
      zn[d] += s * DT_STEP;
    }
    z_next63[b * 3 + 0] = zn[0]; z_next63[b * 3 + 1] = zn[1]; z_next63[b * 3 + 2] = zn[2];
  }
}

// ---------------------------------------------------------------------------
// x_gt[b,t,:] = x[b, min(t+1,63), :]  (float4 copy)
// ---------------------------------------------------------------------------
__global__ void k_xgt(const float* __restrict__ x, float* __restrict__ out)
{
  long idx = (long)blockIdx.x * 256 + threadIdx.x;
  const long total = (long)BATCH * TSEQ * (XDIM / 4);
  if (idx >= total) return;
  long c4 = idx & 1023;
  long bt = idx >> 10;
  int t = (int)(bt & 63);
  int b = (int)(bt >> 6);
  int tp = t + 1; if (tp > 63) tp = 63;
  const float4* src = (const float4*)x;
  float4* dst = (float4*)out;
  dst[((long)b * TSEQ + t) * 1024 + c4] = src[((long)b * TSEQ + tp) * 1024 + c4];
}

// ---------------------------------------------------------------------------

extern "C" void kernel_launch(void* const* d_in, const int* in_sizes, int n_in,
                              void* d_out, int out_size, void* d_ws, size_t ws_size,
                              hipStream_t stream) {
  (void)in_sizes; (void)n_in; (void)out_size; (void)ws_size;
  const float* x      = (const float*)d_in[0];
  const float* u      = (const float*)d_in[1];
  const float* eps_z  = (const float*)d_in[2];
  const float* eps_b  = (const float*)d_in[3];
  const float* enc_w1 = (const float*)d_in[4];
  const float* enc_b1 = (const float*)d_in[5];
  const float* enc_w2 = (const float*)d_in[6];
  const float* enc_b2 = (const float*)d_in[7];
  const float* dec_w1 = (const float*)d_in[8];
  const float* dec_b1 = (const float*)d_in[9];
  const float* dec_w2 = (const float*)d_in[10];
  const float* dec_b2 = (const float*)d_in[11];
  const float* w_ih   = (const float*)d_in[12];
  const float* w_hh   = (const float*)d_in[13];
  const float* b_ih   = (const float*)d_in[14];
  const float* b_hh   = (const float*)d_in[15];
  const float* fc_w   = (const float*)d_in[16];
  const float* fc_b   = (const float*)d_in[17];
  const float* t1w = (const float*)d_in[18], *t1b = (const float*)d_in[19];
  const float* t2w = (const float*)d_in[20], *t2b = (const float*)d_in[21];
  const float* t3w = (const float*)d_in[22], *t3b = (const float*)d_in[23];
  const float* t4w = (const float*)d_in[24], *t4b = (const float*)d_in[25];
  float* out = (float*)d_out;

  // workspace carve-up (256B aligned)
  char* ws = (char*)d_ws;
  size_t off = 0;
  auto take = [&](size_t bytes) -> char* {
    off = (off + 255) & ~(size_t)255;
    char* p = ws + off; off += bytes; return p;
  };
  __bf16* enc_w1t = (__bf16*)take((size_t)EH * XDIM * 2);     // 512 x 4096
  __bf16* dec_w2t = (__bf16*)take((size_t)XDIM * EH * 2);     // 4096 x 512
  __bf16* whh_bf  = (__bf16*)take((size_t)512 * HID * 2);     // 512 x 128
  float*  h_enc   = (float*)take((size_t)8064 * EH * 4);
  float*  zbuf    = (float*)take((size_t)8192 * 3 * 4);
  float*  znext   = (float*)take((size_t)8192 * 3 * 4);
  float*  gates   = (float*)take((size_t)BATCH * 512 * 4);
  float*  hstate  = (float*)take((size_t)BATCH * HID * 4);
  float*  cstate  = (float*)take((size_t)BATCH * HID * 4);
  __bf16* hdec    = (__bf16*)take((size_t)8192 * EH * 2);

  auto nblk = [](int M, int N){ return ((M / 32) * (N / 64) + 7) / 8; };

  // 0) one-time weight prep (bf16 + transpose to N x K)
  k_transpose_bf16<<<(XDIM * EH + 255) / 256, 256, 0, stream>>>(enc_w1, enc_w1t, XDIM, EH);
  k_transpose_bf16<<<(EH * XDIM + 255) / 256, 256, 0, stream>>>(dec_w2, dec_w2t, EH, XDIM);
  k_conv_bf16<<<(512 * HID + 255) / 256, 256, 0, stream>>>(w_hh, whh_bf, 512 * HID);

  // 1) batched encode for steps 0..62  (8064 x 4096 x 512, relu)
  k_wmma_gemm<1, 1, 0, 1><<<nblk(63 * BATCH, EH), 256, 0, stream>>>(
      x, nullptr, 0, XDIM, enc_w1t, enc_b1, h_enc, 0, EH, 63 * BATCH, EH, XDIM);
  k_encode_z<<<(8064 * 3 + 255) / 256, 256, 0, stream>>>(h_enc, enc_w2, enc_b2, eps_z, zbuf, 8064, 0);

  // 2) sequential LSTM scan (steps 0..62), LDS-resident state, WMMA gates
  k_seq<<<1, 256, 0, stream>>>(zbuf, u, eps_b, whh_bf, w_ih, b_ih, b_hh, fc_w, fc_b,
                               t1w, t1b, t2w, t2b, t3w, t3b, t4w, t4b,
                               znext, hstate, cstate, 63);

  // 3) batched decode for steps 0..62  (8064 x 512 x 4096, sigmoid) -> d_out
  k_hdec<<<(8064 * EH + 255) / 256, 256, 0, stream>>>(znext, dec_w1, dec_b1, hdec, 8064);
  k_wmma_gemm<0, 0, 1, 2><<<nblk(63 * BATCH, XDIM), 256, 0, stream>>>(
      nullptr, hdec, 0, EH, dec_w2t, dec_b2, out, 0, XDIM, 63 * BATCH, XDIM, EH);

  // 4) step 63: x_t = x_hat_62 (read back from d_out slice t=62)
  k_wmma_gemm<1, 0, 0, 1><<<nblk(BATCH, EH), 256, 0, stream>>>(
      out, nullptr, 62L * XDIM, 64L * XDIM, enc_w1t, enc_b1, h_enc, 0, EH, BATCH, EH, XDIM);
  k_encode_z<<<(128 * 3 + 255) / 256, 256, 0, stream>>>(h_enc, enc_w2, enc_b2, eps_z,
                                                        zbuf + 8064 * 3, 128, 63);
  k_step63_gates<<<(BATCH * 512 + 255) / 256, 256, 0, stream>>>(zbuf + 8064 * 3, hstate,
                                                                w_ih, w_hh, b_ih, b_hh, gates);
  k_step63_finish<<<1, 512, 0, stream>>>(gates, cstate, zbuf + 8064 * 3, eps_b, fc_w, fc_b,
                                         t1w, t1b, t2w, t2b, t3w, t3b, t4w, t4b,
                                         znext + 8064 * 3);
  k_hdec<<<(128 * EH + 255) / 256, 256, 0, stream>>>(znext + 8064 * 3, dec_w1, dec_b1,
                                                     hdec + (size_t)8064 * EH, 128);
  k_wmma_gemm<0, 0, 1, 2><<<nblk(BATCH, XDIM), 256, 0, stream>>>(
      nullptr, hdec + (size_t)8064 * EH, 0, EH, dec_w2t, dec_b2, out, 63L * XDIM, XDIM,
      BATCH, XDIM, EH);

  // 5) x_gt = x[:, clip(t+1)]
  k_xgt<<<(int)(((long)BATCH * TSEQ * (XDIM / 4) + 255) / 256), 256, 0, stream>>>(x, out + (size_t)BATCH * TSEQ * XDIM);
}
